// SelfTransformerLayer_62139586839041
// MI455X (gfx1250) — matrix operand order repaired and checked by reading, not compiled, not run on hardware
//
#include <hip/hip_runtime.h>

// ---------------------------------------------------------------------------
// MI455X (gfx1250, wave32) implementation of the SelfTransformerLayer.
// All GEMMs (incl. flash attention) run on v_wmma_f32_16x16x32_bf16.
// Attention stages K / V^T chunks into LDS with global_load_async_to_lds_b128.
// ---------------------------------------------------------------------------

#define N_PTS   16384
#define M_PTS   6144
#define NF_IN   32
#define NF      128
#define EPS     1e-5f

typedef __attribute__((ext_vector_type(16))) __bf16          v16bf;
typedef __attribute__((ext_vector_type(16))) unsigned short  v16us;
typedef __attribute__((ext_vector_type(8)))  float           v8f;

// float -> bf16 bits, round-to-nearest-even
__device__ __forceinline__ unsigned short f2bf(float f) {
    unsigned int x = __builtin_bit_cast(unsigned int, f);
    x += 0x7FFFu + ((x >> 16) & 1u);
    return (unsigned short)(x >> 16);
}

// ---- WMMA fragment loaders (contiguous k order, same for A and B) -----------------
// A (16-bit, 16x32 MxK), row-major source, leading dim ld (elements):
//   lanes 0-15  : M = lane,    K = 0..15
//   lanes 16-31 : M = lane-16, K = 16..31
__device__ __forceinline__ v16bf load_a_frag(const unsigned short* p, int ld, int lane) {
    int m  = lane & 15;
    int ko = (lane >> 4) * 16;
    const unsigned short* r = p + (long)m * ld;
    v16us u;
#pragma unroll
    for (int i = 0; i < 16; ++i) u[i] = r[ko + i];
    return __builtin_bit_cast(v16bf, u);
}

// B (16-bit, 32x16 KxN) fragment from a TRANSPOSED source BT[n][k], ld = row
// stride of BT. Element (k, n) = BT[n][k]; per-lane reads are contiguous.
//   lanes 0-15  : N = lane,    K = 0..15
//   lanes 16-31 : N = lane-16, K = 16..31
__device__ __forceinline__ v16bf load_bt_frag(const unsigned short* p, int ld, int lane) {
    int n  = lane & 15;
    int ko = (lane >> 4) * 16;
    const unsigned short* r = p + (long)n * ld + ko;
    v16us u;
#pragma unroll
    for (int i = 0; i < 16; ++i) u[i] = r[i];
    return __builtin_bit_cast(v16bf, u);
}

__device__ __forceinline__ v8f wmma_bf16(v16bf a, v16bf b, v8f c) {
    return __builtin_amdgcn_wmma_f32_16x16x32_bf16(false, a, false, b, (short)0, c,
                                                   false, false);
}

// ---------------------------------------------------------------------------
// generic elementwise kernels
// ---------------------------------------------------------------------------
__global__ void cast_bf16_kernel(const float* __restrict__ in,
                                 unsigned short* __restrict__ out, long n) {
    long i = (long)blockIdx.x * blockDim.x + threadIdx.x;
    long stride = (long)gridDim.x * blockDim.x;
    for (; i < n; i += stride) out[i] = f2bf(in[i]);
}

// out[c*R + r] = bf16(in[r*C + c])   (weight pre-transpose)
__global__ void transpose_cast_kernel(const float* __restrict__ in,
                                      unsigned short* __restrict__ out, int R, int C) {
    long n = (long)R * C;
    long i = (long)blockIdx.x * blockDim.x + threadIdx.x;
    long stride = (long)gridDim.x * blockDim.x;
    for (; i < n; i += stride) {
        int r = (int)(i / C), c = (int)(i % C);
        out[(long)c * R + r] = f2bf(in[i]);
    }
}

__global__ void add_kernel(const float* __restrict__ a, const float* __restrict__ b,
                           float* __restrict__ c, long n) {
    long i = (long)blockIdx.x * blockDim.x + threadIdx.x;
    long stride = (long)gridDim.x * blockDim.x;
    for (; i < n; i += stride) c[i] = a[i] + b[i];
}

// gather rows of X [*,128] by idx, cast to bf16
__global__ void gather_cast_kernel(const float* __restrict__ X,
                                   const int* __restrict__ idx,
                                   unsigned short* __restrict__ out, int rows) {
    int i = blockIdx.x;
    if (i >= rows) return;
    int r = idx[i];
    for (int f = threadIdx.x; f < NF; f += blockDim.x)
        out[(long)i * NF + f] = f2bf(X[(long)r * NF + f]);
}

// ---------------------------------------------------------------------------
// BatchNorm (training-mode) statistics: per-feature mean + population var
// ---------------------------------------------------------------------------
__global__ void bn_stats_kernel(const float* __restrict__ X, int rows,
                                float* __restrict__ mean, float* __restrict__ var) {
    __shared__ float sh0[256];
    __shared__ float sh1[256];
    int f = blockIdx.x;
    float s = 0.f, s2 = 0.f;
    for (int r = threadIdx.x; r < rows; r += blockDim.x) {
        float v = X[(long)r * NF + f];
        s += v; s2 += v * v;
    }
    sh0[threadIdx.x] = s; sh1[threadIdx.x] = s2;
    __syncthreads();
    for (int off = 128; off > 0; off >>= 1) {
        if (threadIdx.x < off) {
            sh0[threadIdx.x] += sh0[threadIdx.x + off];
            sh1[threadIdx.x] += sh1[threadIdx.x + off];
        }
        __syncthreads();
    }
    if (threadIdx.x == 0) {
        float inv = 1.0f / (float)rows;
        float mu = sh0[0] * inv;
        mean[f] = mu;
        var[f]  = sh1[0] * inv - mu * mu;
    }
}

// out = [relu]( (X-mean)*rsqrt(var+eps)*g + b ) [+ res];  fp32 and/or bf16 out
__global__ void bn_apply_kernel(const float* __restrict__ X, const float* __restrict__ res,
                                const float* __restrict__ mean, const float* __restrict__ var,
                                const float* __restrict__ g, const float* __restrict__ b,
                                float* __restrict__ outF, unsigned short* __restrict__ outB,
                                long n, int relu) {
    long i = (long)blockIdx.x * blockDim.x + threadIdx.x;
    long stride = (long)gridDim.x * blockDim.x;
    for (; i < n; i += stride) {
        int f = (int)(i & (NF - 1));
        float v = (X[i] - mean[f]) * __frsqrt_rn(var[f] + EPS) * g[f] + b[f];
        if (relu) v = fmaxf(v, 0.f);
        if (res)  v += res[i];
        if (outF) outF[i] = v;
        if (outB) outB[i] = f2bf(v);
    }
}

// ---------------------------------------------------------------------------
// GEMM: C[Mr x Nc] = A[Mr x K] * B, with B given pre-transposed (BT[n][k]).
// One 16x64 output tile per wave (A-frag reused across 4 column tiles).
// BMODE: 0 = no bf16 out, 1 = row-major bf16, 2 = transposed bf16 (Cb[n][m]).
// ---------------------------------------------------------------------------
template <bool OUTF, int BMODE>
__global__ void gemm_bf16_kernel(const unsigned short* __restrict__ A,
                                 const unsigned short* __restrict__ BT,
                                 float* __restrict__ Cf, unsigned short* __restrict__ Cb,
                                 int Mr, int K, int Nc) {
    int wave = threadIdx.x >> 5;
    int lane = threadIdx.x & 31;
    int tilesN = Nc >> 6;                       // 64-wide wave tiles
    int tile = blockIdx.x * (blockDim.x >> 5) + wave;
    int total = (Mr >> 4) * tilesN;
    if (tile >= total) return;                  // wave-uniform: EXEC stays all-1
    int tm = (tile / tilesN) << 4;
    int tn = (tile % tilesN) << 6;

    v8f acc[4];
#pragma unroll
    for (int c = 0; c < 4; ++c) acc[c] = (v8f){};

    for (int k = 0; k < K; k += 32) {
        v16bf a = load_a_frag(A + (long)tm * K + k, K, lane);
#pragma unroll
        for (int c = 0; c < 4; ++c) {
            v16bf b = load_bt_frag(BT + (long)(tn + c * 16) * K + k, K, lane);
            acc[c] = wmma_bf16(a, b, acc[c]);
        }
    }
    int n  = lane & 15;
    int mo = (lane >> 4) * 8;
#pragma unroll
    for (int c = 0; c < 4; ++c) {
#pragma unroll
        for (int r = 0; r < 8; ++r) {
            float v = acc[c][r];
            long row = tm + mo + r;
            long col = tn + c * 16 + n;
            if constexpr (OUTF)       Cf[row * Nc + col] = v;
            if constexpr (BMODE == 1) Cb[row * Nc + col] = f2bf(v);
            if constexpr (BMODE == 2) Cb[col * (long)Mr + row] = f2bf(v);
        }
    }
}

// ---------------------------------------------------------------------------
// Flash attention: O = softmax(Q K^T) V, d = 128, bf16 in/out.
// One 16-query tile per wave, 8 waves/block; per 32-key chunk the block
// cooperatively stages K (row-major) and V^T (feature-major) into LDS using
// CDNA5 async global->LDS loads, then every wave computes from LDS.
// __launch_bounds__(256, 1): ~170 live VGPRs/wave (Q frags + O accum); allow
// the full register budget so the hot loop has zero scratch spills.
// ---------------------------------------------------------------------------
__global__ void __launch_bounds__(256, 1)
attn_kernel(const unsigned short* __restrict__ Q,
            const unsigned short* __restrict__ Km,
            const unsigned short* __restrict__ VT,   // [NF][Mk]
            unsigned short* __restrict__ Ob,
            int Nq, int Mk) {
    __shared__ unsigned short kbuf[32 * NF];        // 8 KB: [key][feat]
    __shared__ unsigned short vtbuf[NF * 32];       // 8 KB: [feat][key]
    __shared__ unsigned short pbuf[8][16 * 32];     // 8 KB: per-wave P staging

    int wave = threadIdx.x >> 5;
    int lane = threadIdx.x & 31;
    int tid  = threadIdx.x;
    int qt = blockIdx.x * (blockDim.x >> 5) + wave;
    if (qt >= (Nq >> 4)) return;

    const unsigned short* qp = Q + (long)qt * 16 * NF;
    v16bf qa[4];
#pragma unroll
    for (int s = 0; s < 4; ++s) qa[s] = load_a_frag(qp + s * 32, NF, lane);

    v8f o[8];
#pragma unroll
    for (int c = 0; c < 8; ++c) o[c] = (v8f){};
    float mrow[8], lrow[8];
#pragma unroll
    for (int r = 0; r < 8; ++r) { mrow[r] = -1e30f; lrow[r] = 0.f; }

    unsigned short* pl = &pbuf[wave][0];
    int n  = lane & 15;
    int mo = (lane >> 4) * 8;

    unsigned kbase = (unsigned)(unsigned long)(void*)kbuf;
    unsigned vbase = (unsigned)(unsigned long)(void*)vtbuf;

    for (int kb = 0; kb < Mk; kb += 32) {
        // ---- async stage K chunk (8 KB) and V^T chunk (8 KB) into LDS ------
        // 512 16-byte chunks each; 256 threads x 2 iterations.
#pragma unroll
        for (int t = 0; t < 2; ++t) {
            int chunk = tid + t * 256;                         // 0..511
            unsigned lk = kbase + (unsigned)chunk * 16;
            unsigned long gk =
                (unsigned long)(Km + (long)kb * NF) + (unsigned long)chunk * 16;
            asm volatile("global_load_async_to_lds_b128 %0, %1, off"
                         :: "v"(lk), "v"(gk) : "memory");
            int f = chunk >> 2, part = chunk & 3;              // V^T row f, 64B rows
            unsigned lv = vbase + (unsigned)chunk * 16;
            unsigned long gv =
                (unsigned long)(VT + (long)f * Mk + kb + part * 8);
            asm volatile("global_load_async_to_lds_b128 %0, %1, off"
                         :: "v"(lv), "v"(gv) : "memory");
        }
        __builtin_prefetch(Km + (long)(kb + 32) * NF, 0, 0);   // warm next chunk
        asm volatile("s_wait_asynccnt 0" ::: "memory");
        __syncthreads();

        // ---- S = Q * K^T  (two 16x16 tiles: keys kb..+15 and kb+16..+31) ---
        v8f s0 = {}, s1 = {};
#pragma unroll
        for (int s = 0; s < 4; ++s) {
            v16bf b0 = load_bt_frag(kbuf + s * 32, NF, lane);
            v16bf b1 = load_bt_frag(kbuf + 16 * NF + s * 32, NF, lane);
            s0 = wmma_bf16(qa[s], b0, s0);
            s1 = wmma_bf16(qa[s], b1, s1);
        }
        // ---- online softmax over the 32 new scores per row; rescale O ------
#pragma unroll
        for (int r = 0; r < 8; ++r) {
            float mx = fmaxf(s0[r], s1[r]);
#pragma unroll
            for (int off = 1; off < 16; off <<= 1) mx = fmaxf(mx, __shfl_xor(mx, off, 32));
            float nm = fmaxf(mrow[r], mx);
            float scale = __expf(mrow[r] - nm);
            float p0 = __expf(s0[r] - nm);
            float p1 = __expf(s1[r] - nm);
            float ps = p0 + p1;
#pragma unroll
            for (int off = 1; off < 16; off <<= 1) ps += __shfl_xor(ps, off, 32);
            lrow[r] = lrow[r] * scale + ps;
            mrow[r] = nm;
            s0[r] = p0; s1[r] = p1;
#pragma unroll
            for (int c = 0; c < 8; ++c) o[c][r] *= scale;
        }

        // ---- stage P (16x32) through LDS to re-layout C -> A fragment ----
#pragma unroll
        for (int r = 0; r < 8; ++r) {
            pl[(mo + r) * 32 + n]      = f2bf(s0[r]);
            pl[(mo + r) * 32 + 16 + n] = f2bf(s1[r]);
        }
        v16bf pa = load_a_frag(pl, 32, lane);   // same-wave LDS RAW

        // ---- O += P * V  (8 output column tiles, V read from LDS V^T) ----
#pragma unroll
        for (int c = 0; c < 8; ++c) {
            v16bf vfrag = load_bt_frag(vtbuf + c * 16 * 32, 32, lane);
            o[c] = wmma_bf16(pa, vfrag, o[c]);
        }
        __syncthreads();   // all waves done with kbuf/vtbuf before next stage
    }
    // ---- finalize: divide by row sums, store bf16 (row-major [Nq][NF]) ----
#pragma unroll
    for (int c = 0; c < 8; ++c)
#pragma unroll
        for (int r = 0; r < 8; ++r)
            Ob[((long)qt * 16 + mo + r) * NF + c * 16 + n] = f2bf(o[c][r] / lrow[r]);
}

// ---------------------------------------------------------------------------
// host-side orchestration
// ---------------------------------------------------------------------------
extern "C" void kernel_launch(void* const* d_in, const int* in_sizes, int n_in,
                              void* d_out, int out_size, void* d_ws, size_t ws_size,
                              hipStream_t stream) {
    (void)in_sizes; (void)n_in; (void)out_size; (void)ws_size;
    const float* x_feats = (const float*)d_in[0];
    const int*   kv_idx  = (const int*)d_in[1];
    const float* W_p1    = (const float*)d_in[2];
    const float* W_kv    = (const float*)d_in[3];
    const float* g_kv    = (const float*)d_in[4];
    const float* b_kv    = (const float*)d_in[5];
    const float* Wq      = (const float*)d_in[6];
    const float* Wk      = (const float*)d_in[7];
    const float* Wv      = (const float*)d_in[8];
    const float* W_trans = (const float*)d_in[9];
    const float* g_an    = (const float*)d_in[10];
    const float* b_an    = (const float*)d_in[11];
    const float* g_r1    = (const float*)d_in[12];
    const float* b_r1    = (const float*)d_in[13];
    const float* W_r1    = (const float*)d_in[14];
    const float* g_r2    = (const float*)d_in[15];
    const float* b_r2    = (const float*)d_in[16];
    const float* W_r2    = (const float*)d_in[17];
    const float* g_out   = (const float*)d_in[18];
    const float* b_out   = (const float*)d_in[19];
    float* out = (float*)d_out;

    // --- workspace layout ---
    char* ws = (char*)d_ws;
    size_t off = 0;
    auto alloc = [&](size_t bytes) -> char* {
        char* p = ws + off;
        off = (off + bytes + 255) & ~(size_t)255;
        return p;
    };
    unsigned short* xfb   = (unsigned short*)alloc((size_t)N_PTS * NF_IN * 2);
    unsigned short* wp1t  = (unsigned short*)alloc((size_t)NF_IN * NF * 2);
    unsigned short* wkvt  = (unsigned short*)alloc((size_t)NF * NF * 2);
    unsigned short* wqt   = (unsigned short*)alloc((size_t)NF * NF * 2);
    unsigned short* wkt   = (unsigned short*)alloc((size_t)NF * NF * 2);
    unsigned short* wvt   = (unsigned short*)alloc((size_t)NF * NF * 2);
    unsigned short* wtt   = (unsigned short*)alloc((size_t)NF * NF * 2);
    unsigned short* wr1t  = (unsigned short*)alloc((size_t)NF * NF * 2);
    unsigned short* wr2t  = (unsigned short*)alloc((size_t)NF * NF * 2);
    float*          xf32  = (float*)alloc((size_t)N_PTS * NF * 4);
    unsigned short* xb    = (unsigned short*)alloc((size_t)N_PTS * NF * 2);
    unsigned short* qb    = (unsigned short*)alloc((size_t)N_PTS * NF * 2);
    unsigned short* xg    = (unsigned short*)alloc((size_t)M_PTS * NF * 2);
    float*          kvpre = (float*)alloc((size_t)M_PTS * NF * 4);
    unsigned short* kvnb  = (unsigned short*)alloc((size_t)M_PTS * NF * 2);
    unsigned short* kb    = (unsigned short*)alloc((size_t)M_PTS * NF * 2);
    unsigned short* vtb   = (unsigned short*)alloc((size_t)M_PTS * NF * 2);  // [NF][M]
    unsigned short* attnb = (unsigned short*)alloc((size_t)N_PTS * NF * 2);
    float*          tf32  = (float*)alloc((size_t)N_PTS * NF * 4);   // reused: h1f, h2f
    float*          yf32  = (float*)alloc((size_t)N_PTS * NF * 4);
    unsigned short* hb    = (unsigned short*)alloc((size_t)N_PTS * NF * 2); // reused
    float*          mean  = (float*)alloc(NF * 4);
    float*          var   = (float*)alloc(NF * 4);
    float*          zf    = xf32;   // x dead after y is formed
    float*          h1f   = tf32;   // t dead after y is formed
    float*          h2f   = tf32;   // h1 dead after hb2 is formed

    const long nelN = (long)N_PTS * NF;
    const long nelM = (long)M_PTS * NF;
    dim3 blk(256);

    // 1) casts / weight pre-transposes to bf16
    cast_bf16_kernel<<<512, blk, 0, stream>>>(x_feats, xfb, (long)N_PTS * NF_IN);
    transpose_cast_kernel<<<16, blk, 0, stream>>>(W_p1,    wp1t, NF_IN, NF);
    transpose_cast_kernel<<<64, blk, 0, stream>>>(W_kv,    wkvt, NF, NF);
    transpose_cast_kernel<<<64, blk, 0, stream>>>(Wq,      wqt,  NF, NF);
    transpose_cast_kernel<<<64, blk, 0, stream>>>(Wk,      wkt,  NF, NF);
    transpose_cast_kernel<<<64, blk, 0, stream>>>(Wv,      wvt,  NF, NF);
    transpose_cast_kernel<<<64, blk, 0, stream>>>(W_trans, wtt,  NF, NF);
    transpose_cast_kernel<<<64, blk, 0, stream>>>(W_r1,    wr1t, NF, NF);
    transpose_cast_kernel<<<64, blk, 0, stream>>>(W_r2,    wr2t, NF, NF);

    int gN = (N_PTS / 16) * (NF / 64) / 8;   // 256 blocks (16x64 wave tiles)
    int gM = (M_PTS / 16) * (NF / 64) / 8;   // 96 blocks

    // 2) x = x_feats @ W_p1  (fp32 + bf16)
    gemm_bf16_kernel<true, 1><<<gN, blk, 0, stream>>>(xfb, wp1t, xf32, xb,
                                                      N_PTS, NF_IN, NF);
    // 3) q = x @ Wq (bf16)
    gemm_bf16_kernel<false, 1><<<gN, blk, 0, stream>>>(xb, wqt, nullptr, qb,
                                                       N_PTS, NF, NF);
    // 4) gather + 5) kv_pre = x[kv_idx] @ W_kv
    gather_cast_kernel<<<M_PTS, 128, 0, stream>>>(xf32, kv_idx, xg, M_PTS);
    gemm_bf16_kernel<true, 0><<<gM, blk, 0, stream>>>(xg, wkvt, kvpre, nullptr,
                                                      M_PTS, NF, NF);
    // 6) kv = BN(kv_pre)
    bn_stats_kernel<<<NF, blk, 0, stream>>>(kvpre, M_PTS, mean, var);
    bn_apply_kernel<<<1024, blk, 0, stream>>>(kvpre, nullptr, mean, var, g_kv, b_kv,
                                              nullptr, kvnb, nelM, 0);
    // 7) k (row-major), v stored transposed [NF][M] for attention
    gemm_bf16_kernel<false, 1><<<gM, blk, 0, stream>>>(kvnb, wkt, nullptr, kb,
                                                       M_PTS, NF, NF);
    gemm_bf16_kernel<false, 2><<<gM, blk, 0, stream>>>(kvnb, wvt, nullptr, vtb,
                                                       M_PTS, NF, NF);
    // 8) attn = softmax(q k^T) v
    attn_kernel<<<(N_PTS / 16) / 8, blk, 0, stream>>>(qb, kb, vtb, attnb,
                                                      N_PTS, M_PTS);
    // 9) y = x + BN(attn @ W_trans)
    gemm_bf16_kernel<true, 0><<<gN, blk, 0, stream>>>(attnb, wtt, tf32, nullptr,
                                                      N_PTS, NF, NF);
    bn_stats_kernel<<<NF, blk, 0, stream>>>(tf32, N_PTS, mean, var);
    bn_apply_kernel<<<2048, blk, 0, stream>>>(tf32, xf32, mean, var, g_an, b_an,
                                              yf32, nullptr, nelN, 0);
    // 10) h = bnrelu(y) @ W_r1
    bn_stats_kernel<<<NF, blk, 0, stream>>>(yf32, N_PTS, mean, var);
    bn_apply_kernel<<<2048, blk, 0, stream>>>(yf32, nullptr, mean, var, g_r1, b_r1,
                                              nullptr, hb, nelN, 1);
    gemm_bf16_kernel<true, 0><<<gN, blk, 0, stream>>>(hb, wr1t, h1f, nullptr,
                                                      N_PTS, NF, NF);
    // 11) h = bnrelu(h) @ W_r2
    bn_stats_kernel<<<NF, blk, 0, stream>>>(h1f, N_PTS, mean, var);
    bn_apply_kernel<<<2048, blk, 0, stream>>>(h1f, nullptr, mean, var, g_r2, b_r2,
                                              nullptr, hb, nelN, 1);
    gemm_bf16_kernel<true, 0><<<gN, blk, 0, stream>>>(hb, wr2t, h2f, nullptr,
                                                      N_PTS, NF, NF);
    // 12) out = bnrelu(y + h)
    add_kernel<<<2048, blk, 0, stream>>>(yf32, h2f, zf, nelN);
    bn_stats_kernel<<<NF, blk, 0, stream>>>(zf, N_PTS, mean, var);
    bn_apply_kernel<<<2048, blk, 0, stream>>>(zf, nullptr, mean, var, g_out, b_out,
                                              out, nullptr, nelN, 1);
}
